// Model_51616916963609
// MI455X (gfx1250) — matrix operand order repaired and checked by reading, not compiled, not run on hardware
//
#include <hip/hip_runtime.h>

// ---------------- model constants ----------------
#define BB      4
#define SEQL    96
#define PREDL   96
#define ENCIN   17
#define CHEMIN  16
#define PATCHL  16
#define STRIDEP 8
#define PN      12
#define DM      512
#define NH      8
#define HD1     64
#define DLLM    4096
#define LHEADS  32
#define LHD     128
#define INTERD  11008
#define DFF     128
#define PROMPT  128
#define LTOT    141                 // PROMPT + PN + 1
#define ROWSLLM (BB*LTOT)           // 564
#define HEADNF  (DFF*PN)            // 1536
#define EPSF    1e-5f

typedef float        v8f_t  __attribute__((ext_vector_type(8)));
typedef __bf16       v16bf_t __attribute__((ext_vector_type(16)));
typedef unsigned int v8u_t  __attribute__((ext_vector_type(8)));
typedef unsigned int u32;

// fp32 pair -> packed bf16 (round-half-up), single v_perm_b32 for the merge
__device__ __forceinline__ u32 pack_bf16(float a, float b) {
    u32 ua = __float_as_uint(a) + 0x8000u;
    u32 ub = __float_as_uint(b) + 0x8000u;
    // result bytes: {ua.b2, ua.b3, ub.b2, ub.b3}
    return __builtin_amdgcn_perm(ub, ua, 0x07060302u);
}

// ---------------------------------------------------------------------------
// WMMA bf16 GEMM:  C(M,N) = A(M,K) @ W(N,K)^T  [+bias][+=C]
// fp32 operands streamed from HBM, converted to bf16 in-register, staged in LDS.
// Block tile 64(M) x 128(N), K-step 64.  8 waves, each owns 32x32 (2x2 frags),
// 8 x v_wmma_f32_16x16x32_bf16 per wave per K-step.
// Requires N % 128 == 0, K % 64 == 0 (true for all large GEMMs here).
// ---------------------------------------------------------------------------
__global__ __launch_bounds__(256)
void wmma_gemm(const float* __restrict__ A, const float* __restrict__ W,
               const float* __restrict__ bias, float* __restrict__ C,
               int M, int N, int K, int addTo)
{
    // LDS tiles: row = 64 halves + pad -> stride 36 u32 (144B, 16B aligned,
    // 36*r mod 64 distinct for r=0..15 -> conflict-free frag reads)
    __shared__ u32 lA[64 * 36];
    __shared__ u32 lB[128 * 36];

    const int tid  = threadIdx.x;
    const int lane = tid & 31;
    const int wave = tid >> 5;
    const int wm   = wave >> 2;          // 0..1
    const int wn   = wave & 3;           // 0..3
    const int mTile = blockIdx.y * 64;
    const int nTile = blockIdx.x * 128;

    // global->LDS load mapping (K-step 64 floats)
    const int aRow = tid >> 2;           // 0..63
    const int aCol = (tid & 3) << 4;     // 0,16,32,48
    const int bRow = tid >> 1;           // 0..127
    const int bCol = (tid & 1) << 5;     // 0,32

    v8f_t acc[2][2];
    const v8f_t vzero = {0.f,0.f,0.f,0.f,0.f,0.f,0.f,0.f};
    acc[0][0] = vzero; acc[0][1] = vzero; acc[1][0] = vzero; acc[1][1] = vzero;

    const int nK = K >> 6;
    const bool aValid = (mTile + aRow) < M;
    const size_t aBase = (size_t)(mTile + aRow) * K + aCol;
    const size_t bBase = (size_t)(nTile + bRow) * K + bCol;

    for (int kb = 0; kb < nK; ++kb) {
        const int k0 = kb << 6;

        // stage global -> registers (fp32)
        float ar[16];
        if (aValid) {
            const float4* ap = (const float4*)(A + aBase + k0);
            #pragma unroll
            for (int j = 0; j < 4; ++j) {
                float4 a4 = ap[j];
                ar[4*j+0]=a4.x; ar[4*j+1]=a4.y; ar[4*j+2]=a4.z; ar[4*j+3]=a4.w;
            }
        } else {
            #pragma unroll
            for (int j = 0; j < 16; ++j) ar[j] = 0.f;
        }
        float br[32];
        {
            const float4* bp = (const float4*)(W + bBase + k0);
            #pragma unroll
            for (int j = 0; j < 8; ++j) {
                float4 b4 = bp[j];
                br[4*j+0]=b4.x; br[4*j+1]=b4.y; br[4*j+2]=b4.z; br[4*j+3]=b4.w;
            }
        }

        __syncthreads();   // previous tile fully consumed

        // convert + store to LDS (bf16 pairs)
        {
            u32* dst = &lA[aRow * 36 + (aCol >> 1)];
            #pragma unroll
            for (int j = 0; j < 8; ++j) dst[j] = pack_bf16(ar[2*j], ar[2*j+1]);
        }
        {
            u32* dst = &lB[bRow * 36 + (bCol >> 1)];
            #pragma unroll
            for (int j = 0; j < 16; ++j) dst[j] = pack_bf16(br[2*j], br[2*j+1]);
        }

        __syncthreads();   // tile visible to all waves

        // prefetch next K tile into cache while the matrix pipe works
        if (kb + 1 < nK) {
            if (aValid) __builtin_prefetch(A + aBase + k0 + 64, 0, 1);
            __builtin_prefetch(W + bBase + k0 + 64, 0, 1);
        }

        #pragma unroll
        for (int kk = 0; kk < 2; ++kk) {
            // A frag: within the 32-half window, lane<16 holds K{0..7,16..23},
            // lane>=16 holds K{8..15,24..31}
            const int aOffU = kk * 16 + ((lane < 16) ? 0 : 4);
            v16bf_t afrag[2];
            #pragma unroll
            for (int mi = 0; mi < 2; ++mi) {
                const int r = wm * 32 + mi * 16 + (lane & 15);
                const uint4* p = (const uint4*)&lA[r * 36 + aOffU];
                uint4 c0 = p[0];
                uint4 c1 = p[2];                  // +8 u32 = +16 halves
                v8u_t t = {c0.x, c0.y, c0.z, c0.w, c1.x, c1.y, c1.z, c1.w};
                afrag[mi] = __builtin_bit_cast(v16bf_t, t);
            }
            // B frag: lane<16 holds K{0..7,16..23}; lane>=16 holds K{8..15,24..31}
            const int bOffU = kk * 16 + ((lane < 16) ? 0 : 4);
            v16bf_t bfrag[2];
            #pragma unroll
            for (int ni = 0; ni < 2; ++ni) {
                const int r = wn * 32 + ni * 16 + (lane & 15);
                const uint4* p = (const uint4*)&lB[r * 36 + bOffU];
                uint4 c0 = p[0];
                uint4 c1 = p[2];
                v8u_t t = {c0.x, c0.y, c0.z, c0.w, c1.x, c1.y, c1.z, c1.w};
                bfrag[ni] = __builtin_bit_cast(v16bf_t, t);
            }

            #pragma unroll
            for (int mi = 0; mi < 2; ++mi)
                #pragma unroll
                for (int ni = 0; ni < 2; ++ni)
                    acc[mi][ni] = __builtin_amdgcn_wmma_f32_16x16x32_bf16(
                        false, afrag[mi], false, bfrag[ni],
                        (short)0, acc[mi][ni], false, false);
        }
    }

    // epilogue: C layout lanes 0-15 -> M=i, lanes 16-31 -> M=8+i; N = lane&15
    const int mBase = mTile + wm * 32 + ((lane >> 4) << 3);
    const int nBase = nTile + wn * 32 + (lane & 15);
    #pragma unroll
    for (int mi = 0; mi < 2; ++mi) {
        #pragma unroll
        for (int ni = 0; ni < 2; ++ni) {
            const int n = nBase + ni * 16;
            const float bv = bias ? bias[n] : 0.f;
            #pragma unroll
            for (int i = 0; i < 8; ++i) {
                const int m = mBase + mi * 16 + i;
                if (m < M) {
                    const size_t idx = (size_t)m * N + n;
                    float v = acc[mi][ni][i] + bv;
                    C[idx] = addTo ? (C[idx] + v) : v;
                }
            }
        }
    }
}

// ---------------------------------------------------------------------------
// small helpers / non-WMMA kernels
// ---------------------------------------------------------------------------
__device__ __forceinline__ float block_reduce_sum(float v) {
    __shared__ float sb[256];
    const int t = threadIdx.x;
    sb[t] = v; __syncthreads();
    for (int s = 128; s > 0; s >>= 1) { if (t < s) sb[t] += sb[t + s]; __syncthreads(); }
    float r = sb[0]; __syncthreads();
    return r;
}

__global__ void stats_kernel(const float* __restrict__ xe, float* means, float* stdev) {
    int i = blockIdx.x * blockDim.x + threadIdx.x;
    if (i >= BB * ENCIN) return;
    int b = i / ENCIN, c = i % ENCIN;
    float s = 0.f;
    for (int t = 0; t < SEQL; ++t) s += xe[(b * SEQL + t) * ENCIN + c];
    float m = s / SEQL, v = 0.f;
    for (int t = 0; t < SEQL; ++t) { float d = xe[(b * SEQL + t) * ENCIN + c] - m; v += d * d; }
    v /= SEQL;
    means[i] = m;
    stdev[i] = sqrtf(v + EPSF);
}

// chem = circular Conv1d(k=3) over first 16 channels + sinusoidal PE  -> (B,96,512)
__global__ void chem_kernel(const float* __restrict__ xe, const float* __restrict__ w,
                            float* __restrict__ chem) {
    int idx = blockIdx.x * blockDim.x + threadIdx.x;
    if (idx >= BB * SEQL * DM) return;
    int d = idx % DM, t = (idx / DM) % SEQL, b = idx / (DM * SEQL);
    float s = 0.f;
    #pragma unroll
    for (int j = 0; j < 3; ++j) {
        int tt = (t + j + SEQL - 1) % SEQL;
        const float* xr = xe + (b * SEQL + tt) * ENCIN;
        const float* wr = w + d * (CHEMIN * 3) + j;
        for (int ci = 0; ci < CHEMIN; ++ci) s += wr[ci * 3] * xr[ci];
    }
    int i2 = d >> 1;
    float div = __expf((float)(2 * i2) * (-9.210340371976184f / (float)DM));
    float ang = (float)t * div;
    s += (d & 1) ? __cosf(ang) : __sinf(ang);
    chem[idx] = s;
}

// patch + soh embedding -> x (B,13,512)
__global__ void patch_kernel(const float* __restrict__ xe,
                             const float* __restrict__ means, const float* __restrict__ stdev,
                             const float* __restrict__ pw, const float* __restrict__ pb,
                             const float* __restrict__ ppos,
                             const float* __restrict__ sw, const float* __restrict__ sb,
                             float* __restrict__ x) {
    int idx = blockIdx.x * blockDim.x + threadIdx.x;
    if (idx >= BB * (PN + 1) * DM) return;
    int d = idx % DM, p = (idx / DM) % (PN + 1), b = idx / (DM * (PN + 1));
    float mu = means[b * ENCIN + ENCIN - 1], sd = stdev[b * ENCIN + ENCIN - 1];
    float s;
    if (p < PN) {
        s = pb[d] + ppos[p * DM + d];
        #pragma unroll
        for (int j = 0; j < PATCHL; ++j) {
            int tt = p * STRIDEP + j; if (tt > SEQL - 1) tt = SEQL - 1;
            float soh = (xe[(b * SEQL + tt) * ENCIN + ENCIN - 1] - mu) / sd;
            s += soh * pw[d * PATCHL + j];
        }
    } else {
        s = sb[d];
        for (int t = 0; t < SEQL; ++t) {
            float soh = (xe[(b * SEQL + t) * ENCIN + ENCIN - 1] - mu) / sd;
            s += soh * sw[d * SEQL + t];
        }
    }
    x[idx] = s;
}

// generic naive linear for small GEMMs: C(M,N) = A(M,K) @ W(N,K)^T + bias
__global__ void linear_naive(const float* __restrict__ A, const float* __restrict__ W,
                             const float* __restrict__ bias, float* __restrict__ C,
                             int M, int N, int K) {
    int idx = blockIdx.x * blockDim.x + threadIdx.x;
    if (idx >= M * N) return;
    int m = idx / N, n = idx % N;
    const float* a = A + (size_t)m * K;
    const float* w = W + (size_t)n * K;
    float s = 0.f;
    for (int k = 0; k < K; ++k) s += a[k] * w[k];
    if (bias) s += bias[n];
    C[idx] = s;
}

// per-(b,h,q) attention, hd=64, Lk<=96, no mask. block = 128 threads.
__global__ void small_attn(const float* __restrict__ q, const float* __restrict__ k,
                           const float* __restrict__ v, float* __restrict__ o,
                           int Lq, int Lk, float scale) {
    __shared__ float sQ[HD1];
    __shared__ float sS[SEQL];
    __shared__ float sR[2];
    int blk = blockIdx.x;
    int qi = blk % Lq, h = (blk / Lq) % NH, b = blk / (Lq * NH);
    int t = threadIdx.x;
    size_t qrow = ((size_t)(b * Lq + qi)) * DM + h * HD1;
    if (t < HD1) sQ[t] = q[qrow + t];
    __syncthreads();
    for (int kk = t; kk < Lk; kk += 128) {
        size_t krow = ((size_t)(b * Lk + kk)) * DM + h * HD1;
        float d = 0.f;
        for (int i = 0; i < HD1; ++i) d += sQ[i] * k[krow + i];
        sS[kk] = d * scale;
    }
    __syncthreads();
    if (t == 0) { float m = -1e30f; for (int i = 0; i < Lk; ++i) m = fmaxf(m, sS[i]); sR[0] = m; }
    __syncthreads();
    float m = sR[0];
    for (int kk = t; kk < Lk; kk += 128) sS[kk] = __expf(sS[kk] - m);
    __syncthreads();
    if (t == 0) { float s = 0.f; for (int i = 0; i < Lk; ++i) s += sS[i]; sR[1] = s; }
    __syncthreads();
    float inv = 1.f / sR[1];
    if (t < HD1) {
        float a = 0.f;
        for (int kk = 0; kk < Lk; ++kk)
            a += sS[kk] * v[((size_t)(b * Lk + kk)) * DM + h * HD1 + t];
        o[qrow + t] = a * inv;
    }
}

// out = LayerNorm(x + add) * g + b   (D=512, block 256 per row)
__global__ void layernorm_add(const float* __restrict__ x, const float* __restrict__ add,
                              const float* __restrict__ g, const float* __restrict__ bb,
                              float* __restrict__ out, int D) {
    int r = blockIdx.x, t = threadIdx.x;
    float s = 0.f;
    for (int c = t; c < D; c += 256) { float v = x[(size_t)r * D + c] + add[(size_t)r * D + c]; s += v; }
    float mean = block_reduce_sum(s) / D;
    float s2 = 0.f;
    for (int c = t; c < D; c += 256) {
        float v = x[(size_t)r * D + c] + add[(size_t)r * D + c] - mean; s2 += v * v;
    }
    float var = block_reduce_sum(s2) / D;
    float inv = rsqrtf(var + EPSF);
    for (int c = t; c < D; c += 256) {
        float v = x[(size_t)r * D + c] + add[(size_t)r * D + c];
        out[(size_t)r * D + c] = (v - mean) * inv * g[c] + bb[c];
    }
}

// out = x * rsqrt(mean(x^2)+eps) * w   (D=4096, block 256 per row)
__global__ void rmsnorm_kernel(const float* __restrict__ x, const float* __restrict__ w,
                               float* __restrict__ out, int D) {
    int r = blockIdx.x, t = threadIdx.x;
    float s = 0.f;
    for (int c = t; c < D; c += 256) { float v = x[(size_t)r * D + c]; s += v * v; }
    float ms = block_reduce_sum(s) / D;
    float inv = rsqrtf(ms + EPSF);
    for (int c = t; c < D; c += 256)
        out[(size_t)r * D + c] = x[(size_t)r * D + c] * inv * w[c];
}

// h[b,l,:] = (l<128 ? embed[prompt_ids[b,l]] : enc[b*13 + l-128])
__global__ void gather_h(const float* __restrict__ embed, const int* __restrict__ ids,
                         const float* __restrict__ enc, float* __restrict__ h) {
    int idx = blockIdx.x * blockDim.x + threadIdx.x;
    if (idx >= ROWSLLM * DLLM) return;
    int c = idx % DLLM, row = idx / DLLM;
    int b = row / LTOT, l = row % LTOT;
    float v;
    if (l < PROMPT) {
        int tok = ids[b * PROMPT + l];
        v = embed[(size_t)tok * DLLM + c];
    } else {
        v = enc[((size_t)(b * (PN + 1) + (l - PROMPT))) * DLLM + c];
    }
    h[idx] = v;
}

// in-place RoPE on (B,L, H*hd) buffer; each thread handles the (d, d+64) pair
__global__ void rope_kernel(float* __restrict__ t) {
    int idx = blockIdx.x * blockDim.x + threadIdx.x;
    if (idx >= ROWSLLM * LHEADS * (LHD / 2)) return;
    int d = idx % (LHD / 2);
    int h = (idx / (LHD / 2)) % LHEADS;
    int row = idx / ((LHD / 2) * LHEADS);
    int l = row % LTOT;
    size_t base = (size_t)row * DLLM + h * LHD + d;
    float inv = __expf(-(float)(2 * d) / (float)LHD * 9.210340371976184f);
    float ang = (float)l * inv;
    float c = __cosf(ang), s = __sinf(ang);
    float a = t[base], b = t[base + LHD / 2];
    t[base]            = a * c - b * s;
    t[base + LHD / 2]  = b * c + a * s;
}

// causal attention, hd=128, L=141. one 128-thread block per (b,h,q).
__global__ void llm_attn(const float* __restrict__ q, const float* __restrict__ k,
                         const float* __restrict__ v, float* __restrict__ o, float scale) {
    __shared__ float sQ[LHD];
    __shared__ float sS[LTOT + 3];
    __shared__ float sR[2];
    int blk = blockIdx.x;
    int qi = blk % LTOT, h = (blk / LTOT) % LHEADS, b = blk / (LTOT * LHEADS);
    int t = threadIdx.x;
    size_t qrow = ((size_t)(b * LTOT + qi)) * DLLM + h * LHD;
    sQ[t] = q[qrow + t];
    __syncthreads();
    for (int kk = t; kk < LTOT; kk += 128) {
        float d = -1e30f;
        if (kk <= qi) {
            size_t krow = ((size_t)(b * LTOT + kk)) * DLLM + h * LHD;
            d = 0.f;
            for (int i = 0; i < LHD; ++i) d += sQ[i] * k[krow + i];
            d *= scale;
        }
        sS[kk] = d;
    }
    __syncthreads();
    if (t == 0) { float m = -1e30f; for (int i = 0; i <= qi; ++i) m = fmaxf(m, sS[i]); sR[0] = m; }
    __syncthreads();
    float m = sR[0];
    for (int kk = t; kk < LTOT; kk += 128) sS[kk] = (kk <= qi) ? __expf(sS[kk] - m) : 0.f;
    __syncthreads();
    if (t == 0) { float s = 0.f; for (int i = 0; i <= qi; ++i) s += sS[i]; sR[1] = s; }
    __syncthreads();
    float inv = 1.f / sR[1];
    float acc = 0.f;
    for (int kk = 0; kk <= qi; ++kk)
        acc += sS[kk] * v[((size_t)(b * LTOT + kk)) * DLLM + h * LHD + t];
    o[qrow + t] = acc * inv;
}

// g = silu(g) * u
__global__ void silu_mul(float* __restrict__ g, const float* __restrict__ u, int n) {
    int idx = blockIdx.x * blockDim.x + threadIdx.x;
    if (idx >= n) return;
    float gv = g[idx];
    g[idx] = (gv / (1.f + __expf(-gv))) * u[idx];
}

// output head: traj (denorm) + rul
__global__ void head_kernel(const float* __restrict__ hn,
                            const float* __restrict__ hw, const float* __restrict__ hb,
                            const float* __restrict__ rw, const float* __restrict__ rb,
                            const float* __restrict__ means, const float* __restrict__ stdev,
                            float* __restrict__ out) {
    int b = blockIdx.x, t = threadIdx.x;
    if (t < PREDL) {
        float s = 0.f;
        for (int f = 0; f < DFF; ++f) {
            const size_t hbase = ((size_t)(b * LTOT + (LTOT - PN))) * DLLM + f;
            const float* wrow = hw + (size_t)t * HEADNF + f * PN;
            for (int pp = 0; pp < PN; ++pp)
                s += hn[hbase + (size_t)pp * DLLM] * wrow[pp];
        }
        s += hb[t];
        out[b * PREDL + t] = s * stdev[b * ENCIN + ENCIN - 1] + means[b * ENCIN + ENCIN - 1];
    } else if (t == PREDL) {
        float s = 0.f;
        for (int f = 0; f < DFF; ++f) {
            const size_t hbase = ((size_t)(b * LTOT + (LTOT - PN))) * DLLM + f;
            const float* wrow = rw + f * PN;
            for (int pp = 0; pp < PN; ++pp)
                s += hn[hbase + (size_t)pp * DLLM] * wrow[pp];
        }
        out[BB * PREDL + b] = s + rb[0];
    }
}

// ---------------------------------------------------------------------------
extern "C" void kernel_launch(void* const* d_in, const int* in_sizes, int n_in,
                              void* d_out, int out_size, void* d_ws, size_t ws_size,
                              hipStream_t stream) {
    (void)in_sizes; (void)n_in; (void)out_size; (void)ws_size;
    // inputs (setup_inputs dict order, params flattened recursively)
    const float* x_enc   = (const float*)d_in[0];
    const int*   ids     = (const int*)  d_in[4];
    const float* ccw     = (const float*)d_in[5];
    const float* pw      = (const float*)d_in[6];
    const float* pb      = (const float*)d_in[7];
    const float* ppos    = (const float*)d_in[8];
    const float* sw      = (const float*)d_in[9];
    const float* sbv     = (const float*)d_in[10];
    const float* a1wq=(const float*)d_in[11], *a1wk=(const float*)d_in[12], *a1wv=(const float*)d_in[13];
    const float* a1bq=(const float*)d_in[14], *a1bk=(const float*)d_in[15], *a1bv=(const float*)d_in[16];
    const float* a1wo=(const float*)d_in[17], *a1bo=(const float*)d_in[18];
    const float* a2wq=(const float*)d_in[19], *a2wk=(const float*)d_in[20], *a2wv=(const float*)d_in[21];
    const float* a2bq=(const float*)d_in[22], *a2bk=(const float*)d_in[23], *a2bv=(const float*)d_in[24];
    const float* a2wo=(const float*)d_in[25], *a2bo=(const float*)d_in[26];
    const float* ln1g=(const float*)d_in[27], *ln1b=(const float*)d_in[28];
    const float* ln2g=(const float*)d_in[29], *ln2b=(const float*)d_in[30];
    const float* projw=(const float*)d_in[31], *projb=(const float*)d_in[32];
    const float* embed=(const float*)d_in[33];
    const float* lln1=(const float*)d_in[34], *lln2=(const float*)d_in[35];
    const float* lwq=(const float*)d_in[36], *lwk=(const float*)d_in[37], *lwv=(const float*)d_in[38];
    const float* lwo=(const float*)d_in[39], *lwg=(const float*)d_in[40], *lwu=(const float*)d_in[41];
    const float* lwd=(const float*)d_in[42];
    const float* fnorm=(const float*)d_in[43];
    const float* headw=(const float*)d_in[44], *headb=(const float*)d_in[45];
    const float* rulw=(const float*)d_in[46],  *rulb=(const float*)d_in[47];
    float* out = (float*)d_out;

    // workspace carve-up
    float* wsf = (float*)d_ws;
    size_t off = 0;
    auto carve = [&](size_t n) { float* p = wsf + off; off += (n + 63) & ~(size_t)63; return p; };
    float* means = carve(BB * ENCIN);
    float* stdev = carve(BB * ENCIN);
    float* chem  = carve((size_t)BB * SEQL * DM);
    float* x     = carve((size_t)BB * (PN + 1) * DM);
    float* xt    = carve((size_t)BB * (PN + 1) * DM);
    float* att   = carve((size_t)BB * (PN + 1) * DM);
    float* qs    = carve((size_t)BB * (PN + 1) * DM);
    float* ks    = carve((size_t)BB * SEQL * DM);
    float* vs    = carve((size_t)BB * SEQL * DM);
    float* enc   = carve((size_t)BB * (PN + 1) * DLLM);
    float* h     = carve((size_t)ROWSLLM * DLLM);
    float* xn    = carve((size_t)ROWSLLM * DLLM);   // reused as attention output
    float* qb    = carve((size_t)ROWSLLM * DLLM);
    float* kb    = carve((size_t)ROWSLLM * DLLM);
    float* vb    = carve((size_t)ROWSLLM * DLLM);
    float* gb    = carve((size_t)ROWSLLM * INTERD);
    float* ub    = carve((size_t)ROWSLLM * INTERD);

    const int M13 = BB * (PN + 1);   // 52

    // ---- front-end embeddings ----
    stats_kernel<<<1, 128, 0, stream>>>(x_enc, means, stdev);
    chem_kernel<<<(BB * SEQL * DM + 255) / 256, 256, 0, stream>>>(x_enc, ccw, chem);
    patch_kernel<<<(M13 * DM + 255) / 256, 256, 0, stream>>>(x_enc, means, stdev,
                                                             pw, pb, ppos, sw, sbv, x);
    // ---- attn1 (self) ----
    int gSmall = (M13 * DM + 255) / 256;
    linear_naive<<<gSmall, 256, 0, stream>>>(x, a1wq, a1bq, qs, M13, DM, DM);
    linear_naive<<<gSmall, 256, 0, stream>>>(x, a1wk, a1bk, ks, M13, DM, DM);
    linear_naive<<<gSmall, 256, 0, stream>>>(x, a1wv, a1bv, vs, M13, DM, DM);
    small_attn<<<BB * NH * (PN + 1), 128, 0, stream>>>(qs, ks, vs, att, PN + 1, PN + 1, 0.125f);
    linear_naive<<<gSmall, 256, 0, stream>>>(att, a1wo, a1bo, xt, M13, DM, DM);
    layernorm_add<<<M13, 256, 0, stream>>>(x, xt, ln1g, ln1b, x, DM);
    // ---- attn2 (cross with chem) ----
    linear_naive<<<gSmall, 256, 0, stream>>>(x, a2wq, a2bq, qs, M13, DM, DM);
    int gChem = (BB * SEQL * DM + 255) / 256;
    linear_naive<<<gChem, 256, 0, stream>>>(chem, a2wk, a2bk, ks, BB * SEQL, DM, DM);
    linear_naive<<<gChem, 256, 0, stream>>>(chem, a2wv, a2bv, vs, BB * SEQL, DM, DM);
    small_attn<<<BB * NH * (PN + 1), 128, 0, stream>>>(qs, ks, vs, att, PN + 1, SEQL, 0.125f);
    linear_naive<<<gSmall, 256, 0, stream>>>(att, a2wo, a2bo, xt, M13, DM, DM);
    layernorm_add<<<M13, 256, 0, stream>>>(x, xt, ln2g, ln2b, x, DM);
    // ---- project to LLM dim (WMMA) ----
    {
        dim3 g(DLLM / 128, (M13 + 63) / 64);
        wmma_gemm<<<g, 256, 0, stream>>>(x, projw, projb, enc, M13, DLLM, DM, 0);
    }
    gather_h<<<(ROWSLLM * DLLM + 255) / 256, 256, 0, stream>>>(embed, ids, enc, h);
    // ---- LLaMA layer ----
    rmsnorm_kernel<<<ROWSLLM, 256, 0, stream>>>(h, lln1, xn, DLLM);
    {
        dim3 g(DLLM / 128, (ROWSLLM + 63) / 64);
        wmma_gemm<<<g, 256, 0, stream>>>(xn, lwq, nullptr, qb, ROWSLLM, DLLM, DLLM, 0);
        wmma_gemm<<<g, 256, 0, stream>>>(xn, lwk, nullptr, kb, ROWSLLM, DLLM, DLLM, 0);
        wmma_gemm<<<g, 256, 0, stream>>>(xn, lwv, nullptr, vb, ROWSLLM, DLLM, DLLM, 0);
    }
    int gRope = (ROWSLLM * LHEADS * (LHD / 2) + 255) / 256;
    rope_kernel<<<gRope, 256, 0, stream>>>(qb);
    rope_kernel<<<gRope, 256, 0, stream>>>(kb);
    llm_attn<<<BB * LHEADS * LTOT, 128, 0, stream>>>(qb, kb, vb, xn, 0.08838834764831845f);
    {
        dim3 g(DLLM / 128, (ROWSLLM + 63) / 64);
        wmma_gemm<<<g, 256, 0, stream>>>(xn, lwo, nullptr, h, ROWSLLM, DLLM, DLLM, 1);
    }
    rmsnorm_kernel<<<ROWSLLM, 256, 0, stream>>>(h, lln2, xn, DLLM);
    {
        dim3 g(INTERD / 128, (ROWSLLM + 63) / 64);
        wmma_gemm<<<g, 256, 0, stream>>>(xn, lwg, nullptr, gb, ROWSLLM, INTERD, DLLM, 0);
        wmma_gemm<<<g, 256, 0, stream>>>(xn, lwu, nullptr, ub, ROWSLLM, INTERD, DLLM, 0);
    }
    silu_mul<<<((int)((size_t)ROWSLLM * INTERD + 255) / 256), 256, 0, stream>>>(gb, ub, ROWSLLM * INTERD);
    {
        dim3 g(DLLM / 128, (ROWSLLM + 63) / 64);
        wmma_gemm<<<g, 256, 0, stream>>>(gb, lwd, nullptr, h, ROWSLLM, DLLM, INTERD, 1);
    }
    rmsnorm_kernel<<<ROWSLLM, 256, 0, stream>>>(h, fnorm, xn, DLLM);
    // ---- output head ----
    head_kernel<<<BB, 128, 0, stream>>>(xn, headw, headb, rulw, rulb, means, stdev, out);
}
